// Module2_24532853195173
// MI455X (gfx1250) — compile-verified, matrix operand order checked by reading, not run on hardware
//
#include <hip/hip_runtime.h>
#include <hip/hip_bf16.h>

// ---------------------------------------------------------------------------
// Problem constants (match reference setup_inputs)
// ---------------------------------------------------------------------------
#define NSENTS  4096
#define WTRUNC  40
#define NDOCS   256
#define DTRUNC  20
#define NBLOGS  8
#define BTRUNC  40
#define NPOS    50
#define FEAT    400      // 2*H
#define HDIM    200      // GRU hidden
#define GPAD    208      // per-gate dim padded to 13 * 16 WMMA tiles
#define HPAD    224      // hidden K padded to 7 * 32 WMMA k-steps
#define VOCAB   50000
#define EMBD    300

typedef _Float16 v16h __attribute__((ext_vector_type(16)));
typedef _Float16 v8h  __attribute__((ext_vector_type(8)));
typedef float    v8f  __attribute__((ext_vector_type(8)));

__device__ __forceinline__ v8f zero8() {
  v8f z;
#pragma unroll
  for (int i = 0; i < 8; ++i) z[i] = 0.0f;
  return z;
}

// Build a 16-half WMMA fragment from two contiguous 16B chunks.
// Per CDNA5 ISA 16-bit A/B layout: lane l (row/col = l&15, khi = (l>>4)*8)
// holds K = kbase+khi+0..7 in halves 0..7 and K = kbase+khi+16..23 in 8..15.
__device__ __forceinline__ v16h load_frag(const _Float16* p) {
  v8h lo = *(const v8h*)(p);
  v8h hf = *(const v8h*)(p + 16);
  v16h r;
#pragma unroll
  for (int i = 0; i < 8; ++i) { r[i] = lo[i]; r[i + 8] = hf[i]; }
  return r;
}

#define WMMA_F16(A, B, C) \
  __builtin_amdgcn_wmma_f32_16x16x32_f16(false, (A), false, (B), (short)0, (C), false, false)

// ---------------------------------------------------------------------------
// Weight prep: f32 (3H x DIN) row-major -> f16 [gate][GPAD][DINp], zero padded.
// Padded rows/cols are zero so pad lanes of the GRU stay exactly 0.
// ---------------------------------------------------------------------------
__global__ void prep_gru_weights(const float* __restrict__ wih,
                                 const float* __restrict__ whh,
                                 const float* __restrict__ bih,
                                 const float* __restrict__ bhh,
                                 int DIN, int DINp,
                                 _Float16* __restrict__ wihx,
                                 _Float16* __restrict__ whhx,
                                 float* __restrict__ bihx,
                                 float* __restrict__ bhhx) {
  const int stride = gridDim.x * blockDim.x;
  const int t0 = blockIdx.x * blockDim.x + threadIdx.x;
  const int tot1 = 3 * GPAD * DINp;
  for (int i = t0; i < tot1; i += stride) {
    int d = i % DINp; int rem = i / DINp; int gp = rem % GPAD; int q = rem / GPAD;
    float v = (gp < HDIM && d < DIN) ? wih[(size_t)(q * HDIM + gp) * DIN + d] : 0.0f;
    wihx[i] = (_Float16)v;
  }
  const int tot2 = 3 * GPAD * HPAD;
  for (int i = t0; i < tot2; i += stride) {
    int d = i % HPAD; int rem = i / HPAD; int gp = rem % GPAD; int q = rem / GPAD;
    float v = (gp < HDIM && d < HDIM) ? whh[(size_t)(q * HDIM + gp) * HDIM + d] : 0.0f;
    whhx[i] = (_Float16)v;
  }
  const int tot3 = 3 * GPAD;
  for (int i = t0; i < tot3; i += stride) {
    int gp = i % GPAD; int q = i / GPAD;
    bihx[i] = (gp < HDIM) ? bih[q * HDIM + gp] : 0.0f;
    bhhx[i] = (gp < HDIM) ? bhh[q * HDIM + gp] : 0.0f;
  }
}

// sent_lens[s] = count of nonzero word ids (reference: sum(sign(x)))
__global__ void sent_lens_kernel(const int* __restrict__ x, int* __restrict__ lens) {
  int s = blockIdx.x * blockDim.x + threadIdx.x;
  if (s < NSENTS) {
    int c = 0;
#pragma unroll
    for (int t = 0; t < WTRUNC; ++t) c += (x[s * WTRUNC + t] != 0) ? 1 : 0;
    lens[s] = c;
  }
}

// Serial (tiny) segment maps / prefix sums, faithful to np.repeat/concatenate.
__global__ void build_maps(const int* __restrict__ doc_nums,
                           const int* __restrict__ doc_lens,
                           int* __restrict__ blog_of_doc, int* __restrict__ pos_of_doc,
                           int* __restrict__ doc_of_sent, int* __restrict__ pos_of_sent,
                           int* __restrict__ doc_prefix, int* __restrict__ blog_prefix) {
  if (blockIdx.x == 0 && threadIdx.x == 0) {
    int acc = 0;
    for (int b = 0; b < NBLOGS; ++b) {
      blog_prefix[b] = acc;
      int n = doc_nums[b];
      for (int c = 0; c < n && (acc + c) < NDOCS; ++c) {
        blog_of_doc[acc + c] = b;
        pos_of_doc[acc + c] = c;
      }
      acc += n;
    }
    blog_prefix[NBLOGS] = acc;
    int sacc = 0;
    for (int d = 0; d < NDOCS; ++d) {
      doc_prefix[d] = sacc;
      int l = doc_lens[d];
      for (int c = 0; c < l && (sacc + c) < NSENTS; ++c) {
        doc_of_sent[sacc + c] = d;
        pos_of_sent[sacc + c] = c;
      }
      sacc += l;
    }
    doc_prefix[NDOCS] = sacc;
  }
}

// _pad_segments: out[g][c][:] = vec[prefix[g]+c] if c < lens[g] else 0
__global__ void pad_segments_kernel(const float* __restrict__ vec,
                                    const int* __restrict__ lens,
                                    const int* __restrict__ prefix,
                                    int nseg, int trunc,
                                    float* __restrict__ out) {
  const int total = nseg * trunc * FEAT;
  for (int i = blockIdx.x * blockDim.x + threadIdx.x; i < total;
       i += gridDim.x * blockDim.x) {
    int f = i % FEAT; int rem = i / FEAT; int c = rem % trunc; int g = rem / trunc;
    float v = 0.0f;
    if (c < lens[g]) v = vec[(size_t)(prefix[g] + c) * FEAT + f];
    out[i] = v;
  }
}

// ---------------------------------------------------------------------------
// Fused single-direction GRU scan + masked running max.
// One workgroup = 16 sequences, 8 waves, 13 g-tiles (GPAD/16). Branchless
// tile assignment: waves 0..4 own 2 tiles, waves 5..7 own 1 tile processed
// twice (identical work within the wave -> EXEC stays all-ones around every
// WMMA, no saveexec). Each wave holds all three gate accumulators for its
// tiles so the GRU nonlinearity / h-update / running max stay in registers;
// h round-trips through LDS in f16 as the next step's WMMA A operand.
// B loads are software-pipelined at depth ~2 (<=3 fragments live) and the
// k-loops are kept rolled (#pragma unroll 1) so the live set stays bounded —
// round 2 showed that wider batching + full unroll spilled WMMA operands to
// scratch.
// ---------------------------------------------------------------------------
template <int DINP>
__global__ __launch_bounds__(256, 1) void bigru_dir_kernel(
    const float* __restrict__ input,   // (nseq, T, DIN) or nullptr
    const int* __restrict__ ids,       // (nseq, T) word ids if input==nullptr
    const float* __restrict__ embed,   // (V, DIN) used with ids
    const _Float16* __restrict__ wihx, // [3][GPAD][DINP] f16
    const _Float16* __restrict__ whhx, // [3][GPAD][HPAD] f16
    const float* __restrict__ bihx,    // [3][GPAD]
    const float* __restrict__ bhhx,    // [3][GPAD]
    const int* __restrict__ lens,      // (nseq)
    float* __restrict__ outmax,        // (nseq, 2H): write [outOff, outOff+H)
    int outOff, int nseq, int T, int DIN, int reverse) {
  constexpr int KXI = DINP / 32;  // input k-steps (10 or 13)
  constexpr int KXH = HPAD / 32;  // 7

  __shared__ _Float16 lds_ax[16 * DINP];  // staged input tile (f16, K-padded)
  __shared__ _Float16 lds_ah[16 * HPAD];  // hidden state tile (f16)
  __shared__ float s_bih[3 * GPAD];
  __shared__ float s_bhh[3 * GPAD];
  __shared__ int s_len[16];

  const int tid = threadIdx.x;
  const int seq0 = blockIdx.x * 16;

  for (int i = tid; i < 3 * GPAD; i += 256) { s_bih[i] = bihx[i]; s_bhh[i] = bhhx[i]; }
  for (int i = tid; i < 16 * HPAD; i += 256) lds_ah[i] = (_Float16)0.0f;
  if (tid < 16) { int s = seq0 + tid; s_len[tid] = (s < nseq) ? lens[s] : 0; }
  __syncthreads();

  const int wave = tid >> 5;
  const int lane = tid & 31;
  const int nfrag = lane & 15;      // A row m / B col n / C col n
  const int khi = (lane >> 4) * 8;  // K sub-offset for A/B; C row base
  // Branchless: waves 0..4 -> tiles {2w, 2w+1}; waves 5..7 -> {5+w, 5+w} (dup)
  int gts[2];
  gts[0] = (wave < 5) ? (2 * wave) : (5 + wave);
  gts[1] = (wave < 5) ? (2 * wave + 1) : (5 + wave);

  v8f hreg[2], hmax[2];
#pragma unroll
  for (int j = 0; j < 2; ++j) {
    hreg[j] = zero8();
    hmax[j] = zero8();
#pragma unroll
    for (int v = 0; v < 8; ++v) hmax[j][v] = -3.402823466e38f;
  }

  // Hoisted, strength-reduced B base pointers (row of the f16 weight matrix
  // this lane feeds; advance by 32 halves per k-step inside the phases).
  const size_t rowR0 = (size_t)(0 * GPAD + gts[0] * 16 + nfrag);
  const size_t rowR1 = (size_t)(0 * GPAD + gts[1] * 16 + nfrag);
  const size_t rowZ0 = (size_t)(1 * GPAD + gts[0] * 16 + nfrag);
  const size_t rowZ1 = (size_t)(1 * GPAD + gts[1] * 16 + nfrag);
  const size_t rowN0 = (size_t)(2 * GPAD + gts[0] * 16 + nfrag);
  const size_t rowN1 = (size_t)(2 * GPAD + gts[1] * 16 + nfrag);
  const _Float16* ihR0 = wihx + rowR0 * DINP + khi;
  const _Float16* ihR1 = wihx + rowR1 * DINP + khi;
  const _Float16* ihZ0 = wihx + rowZ0 * DINP + khi;
  const _Float16* ihZ1 = wihx + rowZ1 * DINP + khi;
  const _Float16* ihN0 = wihx + rowN0 * DINP + khi;
  const _Float16* ihN1 = wihx + rowN1 * DINP + khi;
  const _Float16* hhR0 = whhx + rowR0 * HPAD + khi;
  const _Float16* hhR1 = whhx + rowR1 * HPAD + khi;
  const _Float16* hhZ0 = whhx + rowZ0 * HPAD + khi;
  const _Float16* hhZ1 = whhx + rowZ1 * HPAD + khi;
  const _Float16* hhN0 = whhx + rowN0 * HPAD + khi;
  const _Float16* hhN1 = whhx + rowN1 * HPAD + khi;

  for (int rt = 0; rt < T; ++rt) {
    // phase 1: stage this step's 16 input rows into LDS as f16 (zero-padded)
    const int tt = reverse ? (T - 1 - rt) : rt;
    for (int i = tid; i < 16 * DINP; i += 256) {
      const int m = i / DINP;
      const int d = i - m * DINP;
      float v = 0.0f;
      const int s = seq0 + m;
      if (s < nseq && d < DIN) {
        if (ids) {
          const int wid = ids[s * T + tt];
          v = embed[(size_t)wid * DIN + d];
        } else {
          v = input[((size_t)s * T + tt) * DIN + d];
        }
      }
      lds_ax[i] = (_Float16)v;
    }
    __syncthreads();

    v8f accR[2], accZ[2], accNX[2], accNH[2];
#pragma unroll
    for (int j = 0; j < 2; ++j) {
      accR[j] = zero8(); accZ[j] = zero8(); accNX[j] = zero8(); accNH[j] = zero8();
    }

    // phase 2: xg = x_t @ wih^T (WMMA f16 -> f32 accum); depth-2 B pipeline
    {
      const _Float16* ap = &lds_ax[nfrag * DINP + khi];
#pragma unroll 1
      for (int ks = 0; ks < KXI; ++ks) {
        const int kb = ks * 32;
        const v16h a = load_frag(ap + kb);
        v16h b0 = load_frag(ihR0 + kb);
        v16h b1 = load_frag(ihR1 + kb);
        accR[0] = WMMA_F16(a, b0, accR[0]);
        b0 = load_frag(ihZ0 + kb);
        accR[1] = WMMA_F16(a, b1, accR[1]);
        b1 = load_frag(ihZ1 + kb);
        accZ[0] = WMMA_F16(a, b0, accZ[0]);
        b0 = load_frag(ihN0 + kb);
        accZ[1] = WMMA_F16(a, b1, accZ[1]);
        b1 = load_frag(ihN1 + kb);
        accNX[0] = WMMA_F16(a, b0, accNX[0]);
        accNX[1] = WMMA_F16(a, b1, accNX[1]);
      }
    }

    // phase 3: gh = h_{t-1} @ whh^T. r/z fold onto accR/accZ; n kept separate
    // (GRU needs xn and r*hn distinct).
    {
      const _Float16* ap = &lds_ah[nfrag * HPAD + khi];
#pragma unroll 1
      for (int ks = 0; ks < KXH; ++ks) {
        const int kb = ks * 32;
        const v16h a = load_frag(ap + kb);
        v16h b0 = load_frag(hhR0 + kb);
        v16h b1 = load_frag(hhR1 + kb);
        accR[0] = WMMA_F16(a, b0, accR[0]);
        b0 = load_frag(hhZ0 + kb);
        accR[1] = WMMA_F16(a, b1, accR[1]);
        b1 = load_frag(hhZ1 + kb);
        accZ[0] = WMMA_F16(a, b0, accZ[0]);
        b0 = load_frag(hhN0 + kb);
        accZ[1] = WMMA_F16(a, b1, accZ[1]);
        b1 = load_frag(hhN1 + kb);
        accNH[0] = WMMA_F16(a, b0, accNH[0]);
        accNH[1] = WMMA_F16(a, b1, accNH[1]);
      }
    }

    __syncthreads();  // all waves done reading lds_ah/lds_ax

    // phase 4: gates, h update, masked running max, f16 h writeback
#pragma unroll
    for (int j = 0; j < 2; ++j) {
      const int g = gts[j] * 16 + nfrag;  // 0..207 (>=200 are zero-pad lanes)
      const float brz0 = s_bih[0 * GPAD + g] + s_bhh[0 * GPAD + g];
      const float brz1 = s_bih[1 * GPAD + g] + s_bhh[1 * GPAD + g];
      const float bxn = s_bih[2 * GPAD + g];
      const float bhn = s_bhh[2 * GPAD + g];
#pragma unroll
      for (int v = 0; v < 8; ++v) {
        const float r = 1.0f / (1.0f + __expf(-(accR[j][v] + brz0)));
        const float z = 1.0f / (1.0f + __expf(-(accZ[j][v] + brz1)));
        const float nn = tanhf(accNX[j][v] + bxn + r * (accNH[j][v] + bhn));
        const float hn = (1.0f - z) * nn + z * hreg[j][v];
        hreg[j][v] = hn;
        const int m = khi + v;
        const int L = s_len[m];
        const bool inc = reverse ? (rt >= T - L) : (rt < L);
        if (inc) hmax[j][v] = fmaxf(hmax[j][v], hn);
        lds_ah[m * HPAD + g] = (_Float16)hn;
      }
    }
    __syncthreads();
  }

  // write this direction's half of the masked max (dup tiles rewrite same val)
#pragma unroll
  for (int j = 0; j < 2; ++j) {
    const int g = gts[j] * 16 + nfrag;
    if (g < HDIM) {
#pragma unroll
      for (int v = 0; v < 8; ++v) {
        const int m = khi + v;
        const int s = seq0 + m;
        if (s < nseq) {
          const float val = (s_len[m] > 0) ? hmax[j][v] : 0.0f;
          outmax[(size_t)s * FEAT + outOff + g] = val;
        }
      }
    }
  }
}

// ---------------------------------------------------------------------------
// Scoring heads (tiny; bandwidth-trivial VALU fp32)
// ---------------------------------------------------------------------------
// sal_doc[b] = doc_salience_w @ blog_vec[b]
__global__ void sal_doc_kernel(const float* __restrict__ W,
                               const float* __restrict__ blog_vec,
                               float* __restrict__ sal) {
  const int b = blockIdx.x;
  const float* bv = blog_vec + (size_t)b * FEAT;
  for (int d = threadIdx.x; d < FEAT; d += blockDim.x) {
    const float* wr = W + (size_t)d * FEAT;
    float acc = 0.0f;
    for (int e = 0; e < FEAT; ++e) acc += wr[e] * bv[e];
    sal[(size_t)b * FEAT + d] = acc;
  }
}

// sal_sent[dd] = Wl @ blog_vec[blog(dd)] + Wr @ doc_vec[dd], W = (400,800)
__global__ void sal_sent_kernel(const float* __restrict__ W,
                                const float* __restrict__ blog_vec,
                                const float* __restrict__ doc_vec,
                                const int* __restrict__ blog_of_doc,
                                float* __restrict__ sal) {
  const int dd = blockIdx.x;
  const float* bv = blog_vec + (size_t)blog_of_doc[dd] * FEAT;
  const float* dv = doc_vec + (size_t)dd * FEAT;
  for (int d = threadIdx.x; d < FEAT; d += blockDim.x) {
    const float* wr = W + (size_t)d * (2 * FEAT);
    float acc = 0.0f;
    for (int e = 0; e < FEAT; ++e) acc += wr[e] * bv[e];
    for (int e = 0; e < FEAT; ++e) acc += wr[FEAT + e] * dv[e];
    sal[(size_t)dd * FEAT + d] = acc;
  }
}

__global__ void doc_pre_kernel(const float* __restrict__ doc_vec,
                               const float* __restrict__ sal_doc,
                               const int* __restrict__ blog_of_doc,
                               const int* __restrict__ pos_of_doc,
                               const float* __restrict__ content_w,
                               const float* __restrict__ pos_emb,
                               const float* __restrict__ pos_w,
                               const float* __restrict__ bias,
                               float* __restrict__ out) {
  const int n = blockIdx.x;
  const int tid = threadIdx.x;
  const float* dv = doc_vec + (size_t)n * FEAT;
  const float* sd = sal_doc + (size_t)blog_of_doc[n] * FEAT;
  float part = 0.0f;
  for (int i = tid; i < FEAT; i += 256) part += dv[i] * (content_w[i] + sd[i]);
  const float* pe = pos_emb + (size_t)pos_of_doc[n] * NPOS;
  for (int i = tid; i < NPOS; i += 256) part += pe[i] * pos_w[i];
  __shared__ float red[256];
  red[tid] = part;
  __syncthreads();
  for (int st = 128; st > 0; st >>= 1) {
    if (tid < st) red[tid] += red[tid + st];
    __syncthreads();
  }
  if (tid == 0) out[n] = red[0] + bias[0];
}

__global__ void sent_pre_kernel(const float* __restrict__ sent_vec,
                                const float* __restrict__ sal_sent,
                                const int* __restrict__ doc_of_sent,
                                const int* __restrict__ pos_of_sent,
                                const int* __restrict__ blog_of_doc,
                                const float* __restrict__ content_w,
                                const float* __restrict__ pos_emb,
                                const float* __restrict__ pos_w,
                                const float* __restrict__ doc_label_w,
                                const float* __restrict__ bias,
                                const float* __restrict__ doc_pre,
                                float* __restrict__ out) {
  const int s = blockIdx.x;
  const int tid = threadIdx.x;
  const int dd = doc_of_sent[s];
  const float* sv = sent_vec + (size_t)s * FEAT;
  const float* ss = sal_sent + (size_t)dd * FEAT;
  float part = 0.0f;
  for (int i = tid; i < FEAT; i += 256) part += sv[i] * (content_w[i] + ss[i]);
  const float* pe = pos_emb + (size_t)pos_of_sent[s] * NPOS;
  for (int i = tid; i < NPOS; i += 256) part += pe[i] * pos_w[i];
  __shared__ float red[256];
  red[tid] = part;
  __syncthreads();
  for (int st = 128; st > 0; st >>= 1) {
    if (tid < st) red[tid] += red[tid + st];
    __syncthreads();
  }
  if (tid == 0) {
    // NOTE: reference indexes doc_pre with blog_of_sent (values 0..7) —
    // replicated faithfully, quirk and all.
    const int blg = blog_of_doc[dd];
    out[s] = red[0] + doc_label_w[0] * doc_pre[blg] + bias[0];
  }
}

// ---------------------------------------------------------------------------
// Host launcher
// ---------------------------------------------------------------------------
extern "C" void kernel_launch(void* const* d_in, const int* in_sizes, int n_in,
                              void* d_out, int out_size, void* d_ws, size_t ws_size,
                              hipStream_t stream) {
  (void)in_sizes; (void)n_in; (void)out_size; (void)ws_size;

  // Inputs in setup_inputs() dict order (params flattened in insertion order).
  const int* x         = (const int*)d_in[0];
  const int* doc_nums  = (const int*)d_in[1];
  const int* doc_lens  = (const int*)d_in[2];
  const float* embed   = (const float*)d_in[3];
  const float* gsrc[3][8];
  for (int L = 0; L < 3; ++L)
    for (int k = 0; k < 8; ++k) gsrc[L][k] = (const float*)d_in[4 + L * 8 + k];
  const float* doc_content_w   = (const float*)d_in[28];
  const float* doc_salience_w  = (const float*)d_in[29];
  const float* doc_pos_emb     = (const float*)d_in[30];
  const float* doc_pos_w       = (const float*)d_in[31];
  const float* doc_bias        = (const float*)d_in[32];
  const float* sent_content_w  = (const float*)d_in[33];
  const float* sent_salience_w = (const float*)d_in[34];
  const float* sent_pos_emb    = (const float*)d_in[35];
  const float* sent_pos_w      = (const float*)d_in[36];
  const float* sent_doc_lbl_w  = (const float*)d_in[37];
  const float* sent_bias       = (const float*)d_in[38];

  // Workspace carve-out (256B aligned slices)
  char* wbase = (char*)d_ws;
  size_t off = 0;
  auto take = [&](size_t bytes) -> void* {
    void* p = wbase + off;
    off = (off + bytes + 255) & ~(size_t)255;
    return p;
  };

  const int DINs[3]  = {EMBD, FEAT, FEAT};
  const int DINps[3] = {320, 416, 416};
  _Float16* wihx[3][2]; _Float16* whhx[3][2];
  float* bihxp[3][2];   float* bhhxp[3][2];
  for (int L = 0; L < 3; ++L)
    for (int d = 0; d < 2; ++d) {
      wihx[L][d]  = (_Float16*)take((size_t)3 * GPAD * DINps[L] * 2);
      whhx[L][d]  = (_Float16*)take((size_t)3 * GPAD * HPAD * 2);
      bihxp[L][d] = (float*)take((size_t)3 * GPAD * 4);
      bhhxp[L][d] = (float*)take((size_t)3 * GPAD * 4);
    }
  int* sentlens     = (int*)take(NSENTS * 4);
  int* blog_of_doc  = (int*)take(NDOCS * 4);
  int* pos_of_doc   = (int*)take(NDOCS * 4);
  int* doc_of_sent  = (int*)take(NSENTS * 4);
  int* pos_of_sent  = (int*)take(NSENTS * 4);
  int* doc_prefix   = (int*)take((NDOCS + 1) * 4);
  int* blog_prefix  = (int*)take((NBLOGS + 8) * 4);
  float* sent_vec   = (float*)take((size_t)NSENTS * FEAT * 4);
  float* spad       = (float*)take((size_t)NDOCS * DTRUNC * FEAT * 4);
  float* doc_vec    = (float*)take((size_t)NDOCS * FEAT * 4);
  float* dpad       = (float*)take((size_t)NBLOGS * BTRUNC * FEAT * 4);
  float* blog_vec   = (float*)take((size_t)NBLOGS * FEAT * 4);
  float* sal_doc    = (float*)take((size_t)NBLOGS * FEAT * 4);
  float* sal_sent   = (float*)take((size_t)NDOCS * FEAT * 4);

  // 1) weight prep (6 direction sets)
  for (int L = 0; L < 3; ++L)
    for (int d = 0; d < 2; ++d)
      prep_gru_weights<<<dim3(1024), dim3(256), 0, stream>>>(
          gsrc[L][d * 4 + 0], gsrc[L][d * 4 + 1], gsrc[L][d * 4 + 2],
          gsrc[L][d * 4 + 3], DINs[L], DINps[L], wihx[L][d], whhx[L][d],
          bihxp[L][d], bhhxp[L][d]);

  // 2) lengths + segment maps
  sent_lens_kernel<<<dim3(NSENTS / 256), dim3(256), 0, stream>>>(x, sentlens);
  build_maps<<<dim3(1), dim3(1), 0, stream>>>(doc_nums, doc_lens, blog_of_doc,
                                              pos_of_doc, doc_of_sent,
                                              pos_of_sent, doc_prefix, blog_prefix);

  // 3) word-level BiGRU (fused embedding gather) -> sent_vec (4096 x 400)
  bigru_dir_kernel<320><<<dim3(NSENTS / 16), dim3(256), 0, stream>>>(
      nullptr, x, embed, wihx[0][0], whhx[0][0], bihxp[0][0], bhhxp[0][0],
      sentlens, sent_vec, 0, NSENTS, WTRUNC, EMBD, 0);
  bigru_dir_kernel<320><<<dim3(NSENTS / 16), dim3(256), 0, stream>>>(
      nullptr, x, embed, wihx[0][1], whhx[0][1], bihxp[0][1], bhhxp[0][1],
      sentlens, sent_vec, HDIM, NSENTS, WTRUNC, EMBD, 1);

  // 4) pad -> sentence-level BiGRU -> doc_vec (256 x 400)
  pad_segments_kernel<<<dim3((NDOCS * DTRUNC * FEAT + 255) / 256), dim3(256), 0,
                        stream>>>(sent_vec, doc_lens, doc_prefix, NDOCS, DTRUNC, spad);
  bigru_dir_kernel<416><<<dim3(NDOCS / 16), dim3(256), 0, stream>>>(
      spad, nullptr, nullptr, wihx[1][0], whhx[1][0], bihxp[1][0], bhhxp[1][0],
      doc_lens, doc_vec, 0, NDOCS, DTRUNC, FEAT, 0);
  bigru_dir_kernel<416><<<dim3(NDOCS / 16), dim3(256), 0, stream>>>(
      spad, nullptr, nullptr, wihx[1][1], whhx[1][1], bihxp[1][1], bhhxp[1][1],
      doc_lens, doc_vec, HDIM, NDOCS, DTRUNC, FEAT, 1);

  // 5) pad -> doc-level BiGRU -> blog_vec (8 x 400)
  pad_segments_kernel<<<dim3((NBLOGS * BTRUNC * FEAT + 255) / 256), dim3(256), 0,
                        stream>>>(doc_vec, doc_nums, blog_prefix, NBLOGS, BTRUNC, dpad);
  bigru_dir_kernel<416><<<dim3(1), dim3(256), 0, stream>>>(
      dpad, nullptr, nullptr, wihx[2][0], whhx[2][0], bihxp[2][0], bhhxp[2][0],
      doc_nums, blog_vec, 0, NBLOGS, BTRUNC, FEAT, 0);
  bigru_dir_kernel<416><<<dim3(1), dim3(256), 0, stream>>>(
      dpad, nullptr, nullptr, wihx[2][1], whhx[2][1], bihxp[2][1], bhhxp[2][1],
      doc_nums, blog_vec, HDIM, NBLOGS, BTRUNC, FEAT, 1);

  // 6) scoring heads
  float* out_doc = (float*)d_out;            // [0, 256)
  float* out_sent = (float*)d_out + NDOCS;   // [256, 4352)
  sal_doc_kernel<<<dim3(NBLOGS), dim3(256), 0, stream>>>(doc_salience_w,
                                                         blog_vec, sal_doc);
  doc_pre_kernel<<<dim3(NDOCS), dim3(256), 0, stream>>>(
      doc_vec, sal_doc, blog_of_doc, pos_of_doc, doc_content_w, doc_pos_emb,
      doc_pos_w, doc_bias, out_doc);
  sal_sent_kernel<<<dim3(NDOCS), dim3(256), 0, stream>>>(
      sent_salience_w, blog_vec, doc_vec, blog_of_doc, sal_sent);
  sent_pre_kernel<<<dim3(NSENTS), dim3(256), 0, stream>>>(
      sent_vec, sal_sent, doc_of_sent, pos_of_sent, blog_of_doc, sent_content_w,
      sent_pos_emb, sent_pos_w, sent_doc_lbl_w, sent_bias, out_doc, out_sent);
}
